// ListenerModel_68289980007138
// MI455X (gfx1250) — matrix operand order repaired and checked by reading, not compiled, not run on hardware
//
#include <hip/hip_runtime.h>
#include <math.h>

// ---- problem dims ----
#define Bb 64
#define Ll 512
#define Ss 6
#define Hh 8
#define EMB 1024
#define HID 512
#define IMG 2048
#define ATT 256
#define ROWS (Bb * Ll)           // 32768
#define SROWS (Bb * Ss)          // 384

typedef __attribute__((ext_vector_type(16))) __bf16 v16bf;
typedef __attribute__((ext_vector_type(8)))  float v8f;
typedef __attribute__((ext_vector_type(4)))  unsigned short us4;
typedef __attribute__((ext_vector_type(8)))  unsigned short us8;
typedef __attribute__((ext_vector_type(16))) unsigned short us16;

__device__ __forceinline__ unsigned short f2b(float f) {
  unsigned int u = __float_as_uint(f);
  u += 0x7FFFu + ((u >> 16) & 1u);          // round-to-nearest-even
  return (unsigned short)(u >> 16);
}
__device__ __forceinline__ float b2f(unsigned short h) {
  return __uint_as_float(((unsigned int)h) << 16);
}
__device__ __forceinline__ v16bf frag_cast(us16 v) {
  return __builtin_bit_cast(v16bf, v);
}
__device__ __forceinline__ v8f wmma_bf16(v16bf a, v16bf b, v8f c) {
  return __builtin_amdgcn_wmma_f32_16x16x32_bf16(false, a, false, b, (short)0, c,
                                                 false, false);
}

// A fragment (16x32) from f32 row-major [., lda], rows m0..m0+15, k block k0.
// lanes 0-15: VGPR0-3 = K 0..7, VGPR4-7 = K 16..23 ; lanes 16-31: K 8..15 / 24..31
__device__ __forceinline__ v16bf load_a_f32(const float* __restrict__ A, int lda,
                                            int m0, int k0, int lane) {
  int r  = lane & 15;
  int ko = (lane < 16) ? 0 : 8;
  const float* p = A + (size_t)(m0 + r) * lda + k0 + ko;
  float4 x0 = *(const float4*)(p);
  float4 x1 = *(const float4*)(p + 4);
  float4 x2 = *(const float4*)(p + 16);
  float4 x3 = *(const float4*)(p + 20);
  us16 u;
  u[0]  = f2b(x0.x); u[1]  = f2b(x0.y); u[2]  = f2b(x0.z); u[3]  = f2b(x0.w);
  u[4]  = f2b(x1.x); u[5]  = f2b(x1.y); u[6]  = f2b(x1.z); u[7]  = f2b(x1.w);
  u[8]  = f2b(x2.x); u[9]  = f2b(x2.y); u[10] = f2b(x2.z); u[11] = f2b(x2.w);
  u[12] = f2b(x3.x); u[13] = f2b(x3.y); u[14] = f2b(x3.z); u[15] = f2b(x3.w);
  return frag_cast(u);
}

// A fragment from a 16 x lda bf16 tile in LDS (row-major, ushort-stored)
__device__ __forceinline__ v16bf load_a_lds(const unsigned short* T, int lda,
                                            int k0, int lane) {
  int r  = lane & 15;
  int ko = (lane < 16) ? 0 : 8;
  const unsigned short* p = T + r * lda + k0 + ko;
  us8 lo = *(const us8*)(p);
  us8 hi = *(const us8*)(p + 16);
  us16 u;
#pragma unroll
  for (int j = 0; j < 8; ++j) { u[j] = lo[j]; u[8 + j] = hi[j]; }
  return frag_cast(u);
}

// B fragment (32x16) from bf16 Wt stored transposed [N][ldk].
// lanes 0-15: col=lane, K 0..7 / 16..23 ; lanes 16-31: col=lane-16, K 8..15 / 24..31
__device__ __forceinline__ v16bf load_b(const unsigned short* __restrict__ Wt,
                                        int ldk, int n0, int k0, int lane) {
  int c  = lane & 15;
  int ko = (lane < 16) ? 0 : 8;
  const unsigned short* p = Wt + (size_t)(n0 + c) * ldk + k0 + ko;
  us8 lo = *(const us8*)(p);
  us8 hi = *(const us8*)(p + 16);
  us16 u;
#pragma unroll
  for (int j = 0; j < 8; ++j) { u[j] = lo[j]; u[8 + j] = hi[j]; }
  return frag_cast(u);
}

// ---------------- kernel 0: weight transpose + f32->bf16 --------------------
__global__ __launch_bounds__(256) void k_transpose_bf16(
    const float* __restrict__ in, unsigned short* __restrict__ out, int K, int N) {
  int idx = blockIdx.x * 256 + threadIdx.x;
  if (idx < K * N) {
    int k = idx / N, n = idx - k * N;
    out[(size_t)n * K + k] = f2b(in[idx]);
  }
}

// ---------------- kernel 1: generic WMMA gemm, f32 A ------------------------
__global__ __launch_bounds__(32) void k_gemm_f32a(
    const float* __restrict__ A, int lda,
    const unsigned short* __restrict__ Wt, int K,
    const float* __restrict__ bias,   // may be null
    float* __restrict__ out, int ldo, int relu) {
  int lane = threadIdx.x;
  int m0 = blockIdx.x * 16, n0 = blockIdx.y * 16;
  v8f acc = {0.f, 0.f, 0.f, 0.f, 0.f, 0.f, 0.f, 0.f};
  for (int k0 = 0; k0 < K; k0 += 32)
    acc = wmma_bf16(load_a_f32(A, lda, m0, k0, lane), load_b(Wt, K, n0, k0, lane), acc);
  int ro = (lane < 16) ? 0 : 8;
  int c  = lane & 15;
  float bv = bias ? bias[n0 + c] : 0.f;
#pragma unroll
  for (int i = 0; i < 8; ++i) {
    float v = acc[i] + bv;
    if (relu) v = v > 0.f ? v : 0.f;
    out[(size_t)(m0 + ro + i) * ldo + n0 + c] = v;
  }
}

// ---------------- kernel 2: fused token pipeline ----------------------------
// LDS arena (64 KB):
//   [0 .. 32K)   Abuf : 16x1024 bf16 reps tile    (stage A only)
//   [0 .. 16K)   AT   : 16x256 f32 tanh buffer    (stage C/D, overlaps Abuf)
//   [32K.. 48K)  T1   : 16x512 bf16               (stage A write, B read)
//   [48K.. 64K)  T2   : 16x512 bf16               (stage B write, C/D read)
__global__ __launch_bounds__(256) void k_fused_tokens(
    const float* __restrict__ reps,                 // [ROWS, EMB]
    const unsigned short* __restrict__ WtEmb,       // [HID][EMB]
    const float* __restrict__ bEmb,
    const unsigned short* __restrict__ WtMmA,       // [HID][HID]
    const float* __restrict__ ctxmm,                // [Bb][HID]
    const float* __restrict__ bMm,
    const unsigned short* __restrict__ WtA1,        // [ATT][HID]
    const float* __restrict__ bA1,
    const float* __restrict__ Wa2,                  // [ATT]
    const float* __restrict__ bA2,                  // [1]
    const unsigned char* __restrict__ masks,        // [ROWS]
    unsigned short* __restrict__ mmbf,              // [ROWS][HID] bf16
    float* __restrict__ scores)                     // [ROWS]
{
  __shared__ __align__(16) unsigned char smem[64 * 1024];
  unsigned short* Abuf = (unsigned short*)smem;                  // 32 KB
  float*          AT   = (float*)smem;                           // 16 KB (overlap)
  unsigned short* T1   = (unsigned short*)(smem + 32 * 1024);    // 16 KB
  unsigned short* T2   = (unsigned short*)(smem + 48 * 1024);    // 16 KB

  int tid  = threadIdx.x;
  int wave = tid >> 5, lane = tid & 31;
  int r0 = blockIdx.x * 16;                 // tile base row
  int b  = r0 >> 9;                         // row / 512 (tiles never span batches)
  int ro = (lane < 16) ? 0 : 8;
  int c  = lane & 15;

  // ---- stage A0: cooperative reps tile -> LDS bf16 (convert each elem once)
#pragma unroll
  for (int it = 0; it < 16; ++it) {
    int v4 = tid + it * 256;                // 0..4095 float4s
    float4 x = *(const float4*)(reps + (size_t)r0 * EMB + (size_t)v4 * 4);
    us4 h;
    h[0] = f2b(x.x); h[1] = f2b(x.y); h[2] = f2b(x.z); h[3] = f2b(x.w);
    *(us4*)(Abuf + v4 * 4) = h;
  }
  __syncthreads();

  // ---- stage A: T1 = relu(reps @ W_emb + b_emb), bf16 into LDS ----
#pragma unroll
  for (int nt = 0; nt < 4; ++nt) {
    int n0 = (wave * 4 + nt) * 16;
    v8f acc = {0.f, 0.f, 0.f, 0.f, 0.f, 0.f, 0.f, 0.f};
    for (int k0 = 0; k0 < EMB; k0 += 32)
      acc = wmma_bf16(load_a_lds(Abuf, EMB, k0, lane),
                      load_b(WtEmb, EMB, n0, k0, lane), acc);
    float bv = bEmb[n0 + c];
#pragma unroll
    for (int i = 0; i < 8; ++i) {
      float v = acc[i] + bv;
      v = v > 0.f ? v : 0.f;
      T1[(ro + i) * HID + n0 + c] = f2b(v);
    }
  }
  __syncthreads();

  // ---- stage B: T2 = relu(T1 @ W_mm_top + ctx_mm[b] + b_mm) ----
#pragma unroll
  for (int nt = 0; nt < 4; ++nt) {
    int n0 = (wave * 4 + nt) * 16;
    v8f acc = {0.f, 0.f, 0.f, 0.f, 0.f, 0.f, 0.f, 0.f};
    for (int k0 = 0; k0 < HID; k0 += 32)
      acc = wmma_bf16(load_a_lds(T1, HID, k0, lane),
                      load_b(WtMmA, HID, n0, k0, lane), acc);
    float add = ctxmm[(size_t)b * HID + n0 + c] + bMm[n0 + c];
#pragma unroll
    for (int i = 0; i < 8; ++i) {
      float v = acc[i] + add;
      v = v > 0.f ? v : 0.f;
      T2[(ro + i) * HID + n0 + c] = f2b(v);
    }
  }
  __syncthreads();   // T2 complete; Abuf lifetime over -> AT region reusable

  // coalesced copy T2 -> global mm_context (bf16)
#pragma unroll
  for (int it = 0; it < 4; ++it) {
    int v8i = tid + it * 256;               // 0..1023, each moves 8 ushorts
    *(us8*)(mmbf + (size_t)r0 * HID + (size_t)v8i * 8) = *(const us8*)(&T2[v8i * 8]);
  }

  // ---- stage C: AT = tanh(T2 @ W_a1 + b_a1) ----
#pragma unroll
  for (int nt = 0; nt < 2; ++nt) {
    int n0 = (wave * 2 + nt) * 16;
    v8f acc = {0.f, 0.f, 0.f, 0.f, 0.f, 0.f, 0.f, 0.f};
    for (int k0 = 0; k0 < HID; k0 += 32)
      acc = wmma_bf16(load_a_lds(T2, HID, k0, lane),
                      load_b(WtA1, HID, n0, k0, lane), acc);
    float bv = bA1[n0 + c];
#pragma unroll
    for (int i = 0; i < 8; ++i)
      AT[(ro + i) * ATT + n0 + c] = tanhf(acc[i] + bv);
  }
  __syncthreads();

  // ---- stage D: scores = AT @ W_a2 + b_a2, masked ----
  if (tid < 16) {
    float s = bA2[0];
    for (int k = 0; k < ATT; ++k) s += AT[tid * ATT + k] * Wa2[k];
    int gr = r0 + tid;
    if (masks[gr]) s = -__builtin_inff();
    scores[gr] = s;
  }
}

// ---------------- kernel 3: softmax over L + attended_hid -------------------
__global__ __launch_bounds__(512) void k_softmax_attend(
    const float* __restrict__ scores,               // [Bb][Ll]
    const unsigned short* __restrict__ mmbf,        // [ROWS][HID]
    float* __restrict__ attended)                   // [Bb][HID]
{
  __shared__ float wls[Ll];
  __shared__ float redm[16];
  __shared__ float reds[16];
  int b = blockIdx.x, tid = threadIdx.x;            // 512 threads

  float s = scores[(size_t)b * Ll + tid];
  float m = s;
  for (int off = 16; off > 0; off >>= 1) m = fmaxf(m, __shfl_xor(m, off, 32));
  if ((tid & 31) == 0) redm[tid >> 5] = m;
  __syncthreads();
  if (tid == 0) {
    float t = redm[0];
    for (int i = 1; i < 16; ++i) t = fmaxf(t, redm[i]);
    redm[0] = t;
  }
  __syncthreads();
  float mx = redm[0];
  float e = expf(s - mx);
  float su = e;
  for (int off = 16; off > 0; off >>= 1) su += __shfl_xor(su, off, 32);
  if ((tid & 31) == 0) reds[tid >> 5] = su;
  __syncthreads();
  if (tid == 0) {
    float t = 0.f;
    for (int i = 0; i < 16; ++i) t += reds[i];
    reds[0] = t;
  }
  __syncthreads();
  wls[tid] = e / reds[0];
  __syncthreads();

  // attended[b, n] = sum_l w[l] * mm[b,l,n]  (tid == n, coalesced over n)
  float acc = 0.f;
  const unsigned short* p = mmbf + (size_t)b * Ll * HID + tid;
  for (int l = 0; l < Ll; ++l) acc += wls[l] * b2f(p[(size_t)l * HID]);
  attended[(size_t)b * HID + tid] = acc;
}

// ---------------- kernel 4: ragged history average + zero d_out -------------
__global__ __launch_bounds__(256) void k_hist_avg(
    const float* __restrict__ hist,                 // [SROWS][Hh][EMB]
    const int* __restrict__ histlen,                // [SROWS]
    float* __restrict__ havg,                       // [SROWS][EMB]
    float* __restrict__ outv)                       // [SROWS] -> zeroed
{
  int bs = blockIdx.x, tid = threadIdx.x;
  int hl = histlen[bs];
  float inv = 1.f / (float)(hl > 0 ? hl : 1);
#pragma unroll
  for (int it = 0; it < 4; ++it) {
    int e = tid + it * 256;
    float s = 0.f;
#pragma unroll
    for (int h = 0; h < Hh; ++h)
      if (h < hl) s += hist[((size_t)bs * Hh + h) * EMB + e];
    havg[(size_t)bs * EMB + e] = s * inv;
  }
  if (tid == 0) outv[bs] = 0.f;
}

// ---------------- kernel 5: sep GEMMs + final dot ---------------------------
__global__ __launch_bounds__(32) void k_sep_dot(
    const float* __restrict__ sepimg,               // [SROWS][IMG]
    const unsigned short* __restrict__ WtSep,       // [HID][IMG]
    const float* __restrict__ bSep,
    const float* __restrict__ havg,                 // [SROWS][EMB]
    const unsigned short* __restrict__ WtEmb,       // [HID][EMB]
    const float* __restrict__ bEmb,
    const int* __restrict__ histlen,                // [SROWS]
    const float* __restrict__ attended,             // [Bb][HID]
    float* __restrict__ outv)                       // [SROWS]
{
  int lane = threadIdx.x;
  int m0 = blockIdx.x * 16, n0 = blockIdx.y * 16;
  v8f acc1 = {0.f, 0.f, 0.f, 0.f, 0.f, 0.f, 0.f, 0.f};
  v8f acc2 = {0.f, 0.f, 0.f, 0.f, 0.f, 0.f, 0.f, 0.f};
  for (int k0 = 0; k0 < IMG; k0 += 32)
    acc1 = wmma_bf16(load_a_f32(sepimg, IMG, m0, k0, lane),
                     load_b(WtSep, IMG, n0, k0, lane), acc1);
  for (int k0 = 0; k0 < EMB; k0 += 32)
    acc2 = wmma_bf16(load_a_f32(havg, EMB, m0, k0, lane),
                     load_b(WtEmb, EMB, n0, k0, lane), acc2);
  int ro = (lane < 16) ? 0 : 8;
  int c  = lane & 15;
  float bs = bSep[n0 + c], be = bEmb[n0 + c];
#pragma unroll
  for (int i = 0; i < 8; ++i) {
    int row = m0 + ro + i;
    float v  = acc1[i] + bs;
    float ha = acc2[i] + be;
    ha = ha > 0.f ? ha : 0.f;
    if (histlen[row] > 0) v += ha;
    float prod = v * attended[(size_t)(row / Ss) * HID + n0 + c];
    for (int off = 8; off > 0; off >>= 1) prod += __shfl_xor(prod, off, 16);
    if ((lane & 15) == 0)
      __hip_atomic_fetch_add(&outv[row], prod, __ATOMIC_RELAXED,
                             __HIP_MEMORY_SCOPE_AGENT);
  }
}

// ---------------- workspace layout (bytes) ----------------------------------
#define OFF_WTVIS 0u                                  // 512*12288*2
#define OFF_WTEMB (OFF_WTVIS + (size_t)HID * (Ss * IMG) * 2)
#define OFF_WTMMA (OFF_WTEMB + (size_t)HID * EMB * 2)
#define OFF_WTMMB (OFF_WTMMA + (size_t)HID * HID * 2)
#define OFF_WTSEP (OFF_WTMMB + (size_t)HID * HID * 2)
#define OFF_WTA1  (OFF_WTSEP + (size_t)HID * IMG * 2)
#define OFF_PC    (OFF_WTA1 + (size_t)ATT * HID * 2)
#define OFF_CTXMM (OFF_PC + (size_t)Bb * HID * 4)
#define OFF_SCORE (OFF_CTXMM + (size_t)Bb * HID * 4)
#define OFF_ATTEN (OFF_SCORE + (size_t)ROWS * 4)
#define OFF_HAVG  (OFF_ATTEN + (size_t)Bb * HID * 4)
#define OFF_MMBF  (OFF_HAVG + (size_t)SROWS * EMB * 4)

extern "C" void kernel_launch(void* const* d_in, const int* in_sizes, int n_in,
                              void* d_out, int out_size, void* d_ws, size_t ws_size,
                              hipStream_t stream) {
  const float* reps    = (const float*)d_in[0];
  const float* sepimg  = (const float*)d_in[1];
  const float* visctx  = (const float*)d_in[2];
  const unsigned char* masks = (const unsigned char*)d_in[3];
  const float* hist    = (const float*)d_in[4];
  const int*   histlen = (const int*)d_in[5];
  const float* W_vis = (const float*)d_in[6];  const float* b_vis = (const float*)d_in[7];
  const float* W_emb = (const float*)d_in[8];  const float* b_emb = (const float*)d_in[9];
  const float* W_mm  = (const float*)d_in[10]; const float* b_mm  = (const float*)d_in[11];
  const float* W_sep = (const float*)d_in[12]; const float* b_sep = (const float*)d_in[13];
  const float* W_a1  = (const float*)d_in[14]; const float* b_a1  = (const float*)d_in[15];
  const float* W_a2  = (const float*)d_in[16]; const float* b_a2  = (const float*)d_in[17];

  char* ws = (char*)d_ws;
  unsigned short* WtVis = (unsigned short*)(ws + OFF_WTVIS);
  unsigned short* WtEmb = (unsigned short*)(ws + OFF_WTEMB);
  unsigned short* WtMmA = (unsigned short*)(ws + OFF_WTMMA);
  unsigned short* WtMmB = (unsigned short*)(ws + OFF_WTMMB);
  unsigned short* WtSep = (unsigned short*)(ws + OFF_WTSEP);
  unsigned short* WtA1  = (unsigned short*)(ws + OFF_WTA1);
  float* pc     = (float*)(ws + OFF_PC);
  float* ctxmm  = (float*)(ws + OFF_CTXMM);
  float* scores = (float*)(ws + OFF_SCORE);
  float* atten  = (float*)(ws + OFF_ATTEN);
  float* havg   = (float*)(ws + OFF_HAVG);
  unsigned short* mmbf = (unsigned short*)(ws + OFF_MMBF);
  float* outv = (float*)d_out;

  // 0) weight conversion (transposed bf16)
  auto tr = [&](const float* in, unsigned short* out, int K, int N) {
    int n = K * N;
    k_transpose_bf16<<<(n + 255) / 256, 256, 0, stream>>>(in, out, K, N);
  };
  tr(W_vis, WtVis, Ss * IMG, HID);
  tr(W_emb, WtEmb, EMB, HID);
  tr(W_mm, WtMmA, HID, HID);                        // rows 0..511 of W_mm
  tr(W_mm + (size_t)HID * HID, WtMmB, HID, HID);    // rows 512..1023
  tr(W_sep, WtSep, IMG, HID);
  tr(W_a1, WtA1, HID, ATT);

  // 1) pc = relu(visual_context @ W_vis + b_vis)   [64, 512]
  k_gemm_f32a<<<dim3(Bb / 16, HID / 16), 32, 0, stream>>>(
      visctx, Ss * IMG, WtVis, Ss * IMG, b_vis, pc, HID, 1);

  // 2) ctx_mm = pc @ W_mm_bot                      [64, 512]
  k_gemm_f32a<<<dim3(Bb / 16, HID / 16), 32, 0, stream>>>(
      pc, HID, WtMmB, HID, nullptr, ctxmm, HID, 0);

  // 3) fused token pipeline
  k_fused_tokens<<<ROWS / 16, 256, 0, stream>>>(
      reps, WtEmb, b_emb, WtMmA, ctxmm, b_mm, WtA1, b_a1, W_a2, b_a2,
      masks, mmbf, scores);

  // 4) softmax + attended_hid
  k_softmax_attend<<<Bb, Ll, 0, stream>>>(scores, mmbf, atten);

  // 5) hist average + zero output
  k_hist_avg<<<SROWS, 256, 0, stream>>>(hist, histlen, havg, outv);

  // 6) sep GEMMs + final dot
  k_sep_dot<<<dim3(SROWS / 16, HID / 16), 32, 0, stream>>>(
      sepimg, WtSep, b_sep, havg, WtEmb, b_emb, histlen, atten, outv);
}
